// Decoder_61177514165016
// MI455X (gfx1250) — matrix operand order repaired and checked by reading, not compiled, not run on hardware
//
#include <hip/hip_runtime.h>
#include <hip/hip_bf16.h>
#include <math.h>

// ---------------------------------------------------------------------------
// Decoder: FC+ReLU -> x = fc + chord/100 -> 3-layer GRU -> masked output
//   B=32, T=512, LATENT=256, HIDDEN=1024, OUT=512, gates 3H=1536
// Strategy:
//   * big input GEMMs (gx = x @ w_ih^T + b_ih) via v_wmma_f32_16x16x32_bf16,
//     4 N-tiles per wave per A-fragment (A reuse, higher WMMA/load ratio)
//   * recurrence: 32 persistent WGs, each owns a 16-col H slice; its 48x512
//     w_hh slice staged into LDS by the Tensor Data Mover (tensor_load_to_lds
//     + s_wait_tensorcnt); per step gh computed with WMMA (2 waves x 3 gate
//     accumulators sharing one h fragment); h double-buffered (bf16 WMMA A,
//     fp32 blend state); one atomic grid barrier per time step.
// ---------------------------------------------------------------------------

typedef __bf16 bf16_t;
typedef __attribute__((ext_vector_type(16))) __bf16 v16bf;
typedef __attribute__((ext_vector_type(8)))  __bf16 v8bf;
typedef __attribute__((ext_vector_type(8)))  float  v8f;
typedef __attribute__((ext_vector_type(4)))  unsigned int u32x4;
typedef __attribute__((ext_vector_type(8)))  int i32x8;
typedef __attribute__((ext_vector_type(4)))  int i32x4;

#define B_    32
#define T_    512
#define LAT_  256
#define HID_  1024
#define OUT_  512
#define G_    1536            // 3*OUT
#define M_    (B_ * T_)       // 16384

#define SCAN_BLOCKS  32       // one per 16-wide H slice
#define SCAN_THREADS 64       // 2 waves: m-tiles 0..1, each does r/z/n gates

// ---------------------------------------------------------------------------
// Helpers
// ---------------------------------------------------------------------------
__device__ __forceinline__ float sigmoidf_(float x) {
    return 1.0f / (1.0f + expf(-x));
}

// A fragment (16x32 bf16, row major, ld = K):
// lanes 0-15: row m0+lane, K {k..k+15};
// lanes 16-31: row m0+lane-16, K {k+16..k+31}.
__device__ __forceinline__ v16bf load_a_frag(const bf16_t* __restrict__ base_row,
                                             int k) {
    return *(const v16bf*)(base_row + k);
}

// TDM: load a [rows x cols] bf16 tile (row stride in elements) into LDS.
// Descriptor layout per CDNA5 ISA 8.3/8.4 (D# group0 + group1, groups 2/3 = 0).
__device__ __forceinline__ void tdm_load_tile_bf16(const bf16_t* gsrc,
                                                   unsigned lds_byte_off,
                                                   int rows, int cols,
                                                   int row_stride_elems) {
    unsigned long long ga = (unsigned long long)(size_t)gsrc;
    u32x4 g0;
    g0[0] = 1u;                                          // count=1, user D#
    g0[1] = lds_byte_off;                                // lds_addr
    g0[2] = (unsigned)(ga & 0xFFFFFFFFu);                // global_addr[31:0]
    g0[3] = (unsigned)((ga >> 32) & 0x01FFFFFFu)         // global_addr[56:32]
            | (2u << 30);                                // type=2 (image)
    i32x8 g1;
    g1[0] = (1 << 16);                                   // data_size=1 (2B)
    g1[1] = (int)((unsigned)(cols & 0xFFFF) << 16);      // tensor_dim0 lo16
    g1[2] = (int)(((unsigned)(cols >> 16) & 0xFFFF)      // tensor_dim0 hi16
            | ((unsigned)(rows & 0xFFFF) << 16));        // tensor_dim1 lo16
    g1[3] = (int)(((unsigned)(rows >> 16) & 0xFFFF)      // tensor_dim1 hi16
            | ((unsigned)(cols & 0xFFFF) << 16));        // tile_dim0
    g1[4] = rows & 0xFFFF;                               // tile_dim1 (tile_dim2=0)
    g1[5] = row_stride_elems;                            // tensor_dim0_stride
    g1[6] = 0;
    g1[7] = 0;
    i32x4 z4 = {0, 0, 0, 0};
#if __clang_major__ >= 23
    i32x8 z8 = {0, 0, 0, 0, 0, 0, 0, 0};
    __builtin_amdgcn_tensor_load_to_lds(g0, g1, z4, z4, z8, 0);
#else
    __builtin_amdgcn_tensor_load_to_lds(g0, g1, z4, z4, 0);
#endif
}

// ---------------------------------------------------------------------------
// Kernel 1: FC + ReLU   out[b,o] = relu(z[b,:] . fc_w[o,:] + fc_b[o])
// ---------------------------------------------------------------------------
__global__ void k_fc_relu(const float* __restrict__ z,
                          const float* __restrict__ w,
                          const float* __restrict__ bias,
                          float* __restrict__ out) {
    int i = blockIdx.x * blockDim.x + threadIdx.x;     // 32*1024
    if (i >= B_ * HID_) return;
    int b = i >> 10, o = i & (HID_ - 1);
    float acc = bias[o];
    const float* zr = z + b * LAT_;
    const float* wr = w + o * LAT_;
    for (int k = 0; k < LAT_; ++k) acc = fmaf(zr[k], wr[k], acc);
    out[i] = fmaxf(acc, 0.0f);
}

// ---------------------------------------------------------------------------
// Kernel 2: fp32 -> bf16 convert
// ---------------------------------------------------------------------------
__global__ void k_cvt_bf16(const float* __restrict__ src,
                           bf16_t* __restrict__ dst, int n) {
    int i = blockIdx.x * blockDim.x + threadIdx.x;
    if (i < n) dst[i] = (bf16_t)src[i];
}

// ---------------------------------------------------------------------------
// Kernel 3: x0[b,t,i] = bf16(fc[b,i] + chord[b,t,i] * 0.01)
// ---------------------------------------------------------------------------
__global__ void k_prep_x(const float* __restrict__ fc,
                         const float* __restrict__ chord,
                         bf16_t* __restrict__ x0) {
    int i = blockIdx.x * blockDim.x + threadIdx.x;     // B*T*HID
    if (i >= B_ * T_ * HID_) return;
    int b = i / (T_ * HID_);
    int hcol = i & (HID_ - 1);
    x0[i] = (bf16_t)(fc[b * HID_ + hcol] + chord[i] * 0.01f);
}

// ---------------------------------------------------------------------------
// Kernel 4: WMMA GEMM  C[M,N] = A[M,K](bf16) * W[N,K]^T(bf16) + bias[N]
// One 16x64 job per wave: 4 N-tiles share one A fragment per K chunk.
// ---------------------------------------------------------------------------
__global__ void k_gemm_gx(const bf16_t* __restrict__ A,
                          const bf16_t* __restrict__ W,
                          const float* __restrict__ bias,
                          float* __restrict__ C,
                          int M, int N, int K) {
    int wave = threadIdx.x >> 5;
    int lane = threadIdx.x & 31;
    int job  = blockIdx.x * 8 + wave;
    int nq   = N >> 6;                 // quad-tile columns (24 for N=1536)
    int mt = job / nq, q = job % nq;
    if ((mt << 4) >= M) return;
    int n0 = q << 6;

    int r = lane & 15, half = lane >> 4;
    const bf16_t* ap = A + (size_t)((mt << 4) + r) * K + half * 16;
    const bf16_t* bp = W + (size_t)(n0 + r) * K + half * 16;
    const size_t bstep = (size_t)16 * K;               // next N tile of W

    v8f acc0 = {}, acc1 = {}, acc2 = {}, acc3 = {};
    for (int k = 0; k < K; k += 32) {
        __builtin_prefetch(ap + k + 256, 0, 1);        // global_prefetch_b8
        v16bf a = load_a_frag(ap, k);
        v16bf b0 = *(const v16bf*)(bp + k);
        v16bf b1 = *(const v16bf*)(bp + bstep + k);
        v16bf b2 = *(const v16bf*)(bp + 2 * bstep + k);
        v16bf b3 = *(const v16bf*)(bp + 3 * bstep + k);
        acc0 = __builtin_amdgcn_wmma_f32_16x16x32_bf16(false, a, false, b0, (short)0, acc0, false, false);
        acc1 = __builtin_amdgcn_wmma_f32_16x16x32_bf16(false, a, false, b1, (short)0, acc1, false, false);
        acc2 = __builtin_amdgcn_wmma_f32_16x16x32_bf16(false, a, false, b2, (short)0, acc2, false, false);
        acc3 = __builtin_amdgcn_wmma_f32_16x16x32_bf16(false, a, false, b3, (short)0, acc3, false, false);
    }
    int row0 = (mt << 4) + half * 8;
    v8f accs[4] = {acc0, acc1, acc2, acc3};
#pragma unroll
    for (int j = 0; j < 4; ++j) {
        int col = n0 + j * 16 + r;
        float bc = bias[col];
#pragma unroll
        for (int v = 0; v < 8; ++v)
            C[(size_t)(row0 + v) * N + col] = accs[j][v] + bc;
    }
}

// ---------------------------------------------------------------------------
// Grid barrier over SCAN_BLOCKS blocks (counter pre-zeroed by host memset)
// ---------------------------------------------------------------------------
__device__ __forceinline__ void grid_sync(unsigned* cnt, unsigned target) {
    __syncthreads();
    if (threadIdx.x == 0) {
        __threadfence();
        atomicAdd(cnt, 1u);
        while (__hip_atomic_load(cnt, __ATOMIC_ACQUIRE,
                                 __HIP_MEMORY_SCOPE_AGENT) < target) {
            __builtin_amdgcn_s_sleep(1);
        }
    }
    __syncthreads();
}

// ---------------------------------------------------------------------------
// Kernel 5: GRU scan.  Block s owns H columns [s*16, s*16+16).
//  LDS holds w_hh rows for the 3 gates of this slice: lds row g (0..47)
//  = global row (g/16)*OUT_ + s*16 + (g%16), staged by TDM (3 tile loads,
//  each 16 consecutive rows x 512 cols).
//  Per step: gh = h(t-1) @ wsl^T via WMMA (2 waves, 3 gate accs each),
//  fp32 gate math, publish h slice into the other h buffer, grid barrier.
// ---------------------------------------------------------------------------
__global__ __launch_bounds__(SCAN_THREADS)
void k_gru_scan(const float* __restrict__ gx,         // [M_, G_] (has b_ih)
                const bf16_t* __restrict__ whh,       // [G_, OUT_] bf16
                const float* __restrict__ bhh,        // [G_]
                float* __restrict__ h_f32,            // [B_, OUT_]
                bf16_t* __restrict__ h_bf,            // [2][B_, OUT_]
                bf16_t* __restrict__ x_next,          // [M_, OUT_] or null
                float* __restrict__ y_out,            // [M_, OUT_] or null
                const int* __restrict__ seq_lens,     // [B_]
                unsigned* __restrict__ cnt) {
    __shared__ __align__(32) bf16_t wsl[48 * OUT_];   // 48 KB
    __shared__ float gh_s[B_ * 48];                   // 6 KB

    const int s    = blockIdx.x;                      // slice 0..31
    const int tid  = threadIdx.x;
    const int wave = tid >> 5, lane = tid & 31;

    // ---- stage w_hh slice into LDS with the Tensor Data Mover -------------
    if (wave == 0) {
#pragma unroll
        for (int gate = 0; gate < 3; ++gate) {
            const bf16_t* src = whh
                + (size_t)(gate * OUT_ + s * 16) * OUT_;     // 16 rows x 512
            unsigned lds_off = (unsigned)(size_t)(&wsl[gate * 16 * OUT_]);
            tdm_load_tile_bf16(src, lds_off, 16, OUT_, OUT_);
        }
        __builtin_amdgcn_s_wait_tensorcnt(0);
    }
    __syncthreads();

    const int mt = wave;                              // 0,1 -> batch rows
    const int r = lane & 15, half = lane >> 4;
    const bf16_t* bp0 = &wsl[(     r) * OUT_ + half * 16];
    const bf16_t* bp1 = &wsl[(16 + r) * OUT_ + half * 16];
    const bf16_t* bp2 = &wsl[(32 + r) * OUT_ + half * 16];

    unsigned phase = 0;
    for (int t = 0; t < T_; ++t) {
        // ---- gh tiles: A = h(t-1) from buffer t&1, 3 gates share A --------
        const bf16_t* ap = h_bf + (size_t)(t & 1) * (B_ * OUT_)
                               + (size_t)(mt * 16 + r) * OUT_ + half * 16;
        v8f acc0 = {}, acc1 = {}, acc2 = {};
        for (int k = 0; k < OUT_; k += 32) {
            v16bf a  = load_a_frag(ap, k);
            v16bf b0 = *(const v16bf*)(bp0 + k);      // LDS ds_load_b128
            v16bf b1 = *(const v16bf*)(bp1 + k);
            v16bf b2 = *(const v16bf*)(bp2 + k);
            acc0 = __builtin_amdgcn_wmma_f32_16x16x32_bf16(false, a, false, b0, (short)0, acc0, false, false);
            acc1 = __builtin_amdgcn_wmma_f32_16x16x32_bf16(false, a, false, b1, (short)0, acc1, false, false);
            acc2 = __builtin_amdgcn_wmma_f32_16x16x32_bf16(false, a, false, b2, (short)0, acc2, false, false);
        }
#pragma unroll
        for (int v = 0; v < 8; ++v) {
            int row = (mt * 16 + half * 8 + v) * 48;
            gh_s[row +      r] = acc0[v];
            gh_s[row + 16 + r] = acc1[v];
            gh_s[row + 32 + r] = acc2[v];
        }
        __syncthreads();

        // ---- gate math + state update (fp32) ------------------------------
        for (int idx = tid; idx < B_ * 16; idx += SCAN_THREADS) {
            int m = idx >> 4, j = idx & 15;
            int jg = s * 16 + j;
            float hr = gh_s[m * 48 +      j] + bhh[           jg];
            float hz = gh_s[m * 48 + 16 + j] + bhh[OUT_     + jg];
            float hn = gh_s[m * 48 + 32 + j] + bhh[2 * OUT_ + jg];
            const float* gxr = gx + (size_t)(m * T_ + t) * G_;
            float xr = gxr[jg], xz = gxr[OUT_ + jg], xn = gxr[2 * OUT_ + jg];
            float rg = sigmoidf_(xr + hr);
            float zg = sigmoidf_(xz + hz);
            float ng = tanhf(xn + rg * hn);
            float hold = h_f32[m * OUT_ + jg];
            float hnew = (1.0f - zg) * ng + zg * hold;
            h_f32[m * OUT_ + jg] = hnew;
            // publish into the *other* h buffer (read at step t+1)
            h_bf[(size_t)((t + 1) & 1) * (B_ * OUT_) + m * OUT_ + jg]
                = (bf16_t)hnew;
            size_t oidx = (size_t)(m * T_ + t) * OUT_ + jg;
            if (x_next) x_next[oidx] = (bf16_t)hnew;
            else        y_out[oidx] = (t < seq_lens[m]) ? hnew : 0.0f;
        }
        __threadfence();
        ++phase;
        grid_sync(cnt, phase * SCAN_BLOCKS);
    }
}

// ---------------------------------------------------------------------------
// Host side
// ---------------------------------------------------------------------------
extern "C" void kernel_launch(void* const* d_in, const int* in_sizes, int n_in,
                              void* d_out, int out_size, void* d_ws, size_t ws_size,
                              hipStream_t stream) {
    (void)in_sizes; (void)n_in; (void)out_size; (void)ws_size;
    const float* z     = (const float*)d_in[0];
    const int*   seq   = (const int*)  d_in[1];
    const float* chord = (const float*)d_in[2];
    const float* fc_w  = (const float*)d_in[3];
    const float* fc_b  = (const float*)d_in[4];
    const float* w_ih[3] = {(const float*)d_in[5],  (const float*)d_in[9],  (const float*)d_in[13]};
    const float* w_hh[3] = {(const float*)d_in[6],  (const float*)d_in[10], (const float*)d_in[14]};
    const float* b_ih[3] = {(const float*)d_in[7],  (const float*)d_in[11], (const float*)d_in[15]};
    const float* b_hh[3] = {(const float*)d_in[8],  (const float*)d_in[12], (const float*)d_in[16]};

    char* ws = (char*)d_ws;
    size_t cur = 0;
    auto carve = [&](size_t bytes) -> char* {
        char* p = ws + cur;
        cur += (bytes + 255) & ~(size_t)255;
        return p;
    };
    unsigned* cnt    = (unsigned*)carve(256);
    float*    hfc    = (float*)   carve((size_t)B_ * HID_ * 4);
    float*    h_f32  = (float*)   carve((size_t)B_ * OUT_ * 4);
    bf16_t*   h_bf   = (bf16_t*)  carve((size_t)2 * B_ * OUT_ * 2);
    bf16_t*   x0     = (bf16_t*)  carve((size_t)M_ * HID_ * 2);
    bf16_t*   xA     = (bf16_t*)  carve((size_t)M_ * OUT_ * 2);
    bf16_t*   xB     = (bf16_t*)  carve((size_t)M_ * OUT_ * 2);
    float*    gx     = (float*)   carve((size_t)M_ * G_ * 4);
    bf16_t*   wih_bf[3], *whh_bf[3];
    const int wih_n[3] = {G_ * HID_, G_ * OUT_, G_ * OUT_};
    for (int l = 0; l < 3; ++l) {
        wih_bf[l] = (bf16_t*)carve((size_t)wih_n[l] * 2);
        whh_bf[l] = (bf16_t*)carve((size_t)G_ * OUT_ * 2);
    }

    // FC + ReLU
    k_fc_relu<<<(B_ * HID_ + 255) / 256, 256, 0, stream>>>(z, fc_w, fc_b, hfc);

    // Weight converts to bf16
    for (int l = 0; l < 3; ++l) {
        k_cvt_bf16<<<(wih_n[l] + 255) / 256, 256, 0, stream>>>(w_ih[l], wih_bf[l], wih_n[l]);
        k_cvt_bf16<<<(G_ * OUT_ + 255) / 256, 256, 0, stream>>>(w_hh[l], whh_bf[l], G_ * OUT_);
    }

    // x0 = relu(fc) + chord/100  (bf16)
    k_prep_x<<<(B_ * T_ * HID_ + 255) / 256, 256, 0, stream>>>(hfc, chord, x0);

    const bf16_t* layer_in[3]  = {x0, xA, xB};
    bf16_t*       layer_out[3] = {xA, xB, nullptr};
    const int     layer_k[3]   = {HID_, OUT_, OUT_};

    for (int l = 0; l < 3; ++l) {
        // gx = x @ w_ih^T + b_ih   (WMMA, 16x64 job per wave)
        int jobs   = (M_ / 16) * (G_ / 64);
        int blocks = jobs / 8;
        k_gemm_gx<<<blocks, 256, 0, stream>>>(layer_in[l], wih_bf[l], b_ih[l],
                                              gx, M_, G_, layer_k[l]);
        // reset scan state
        hipMemsetAsync(cnt,   0, 256, stream);
        hipMemsetAsync(h_f32, 0, (size_t)B_ * OUT_ * 4, stream);
        hipMemsetAsync(h_bf,  0, (size_t)2 * B_ * OUT_ * 2, stream);
        // sequential scan
        k_gru_scan<<<SCAN_BLOCKS, SCAN_THREADS, 0, stream>>>(
            gx, whh_bf[l], b_hh[l], h_f32, h_bf,
            layer_out[l], (l == 2) ? (float*)d_out : nullptr, seq, cnt);
    }
}
